// PointerNet_36893769073370
// MI455X (gfx1250) — matrix operand order repaired and checked by reading, not compiled, not run on hardware
//
#include <hip/hip_runtime.h>

typedef __attribute__((ext_vector_type(16))) _Float16 v16h;
typedef __attribute__((ext_vector_type(8)))  _Float16 v8h;
typedef __attribute__((ext_vector_type(8)))  float    v8f;

#define Hh   512
#define Bb   32
#define Ee   128
#define Tt   128
#define H4   2048

// ---------------------------------------------------------------------------
// helpers
// ---------------------------------------------------------------------------
static __device__ __forceinline__ v16h load_a_frag(const _Float16* rowbase, int koff) {
  // rowbase = A + m*lda + 8*(lane>>4); elements 0..7 <- K=koff+kh..+7, 8..15 <- +16
  v8h lo = *(const v8h*)(rowbase + koff);
  v8h hi = *(const v8h*)(rowbase + koff + 16);
  return __builtin_shufflevector(lo, hi, 0,1,2,3,4,5,6,7,8,9,10,11,12,13,14,15);
}

static __device__ __forceinline__ float sigm(float x) {
  return 1.0f / (1.0f + __expf(-x));
}

// ---------------------------------------------------------------------------
// elementwise embedding: enc_in[b,e,k] = endpoints[b,e]*ep_w[k]+ep_b[k]+L[b]*Lw[k]+Lb[k]
// ---------------------------------------------------------------------------
__global__ void k_embed(const float* __restrict__ endpoints, const float* __restrict__ L,
                        const float* __restrict__ ep_w, const float* __restrict__ ep_b,
                        const float* __restrict__ Lw, const float* __restrict__ Lb,
                        _Float16* __restrict__ enc_in) {
  size_t tid = (size_t)blockIdx.x * 256 + threadIdx.x;   // B*E*H elements
  int k = (int)(tid & (Hh - 1));
  size_t be = tid >> 9;
  int b = (int)(be >> 7);
  float v = endpoints[be] * ep_w[k] + ep_b[k] + L[b] * Lw[k] + Lb[k];
  enc_in[tid] = (_Float16)v;
}

// ---------------------------------------------------------------------------
// pack W (N x Ksrc row-major f32, used as out[.,n] = sum_k a[k]*W[n,k]) into
// per-lane WMMA B-fragment layout:
//   dst[((nt*KT_total + kt_base+ktl)*32 + lane)*16 + j] = W[n, k]
//   n = nt*16 + (lane&15);  k = ktl*32 + (lane>>4)*8 + (j&7) + (j>>3)*16
// ---------------------------------------------------------------------------
__global__ void k_pack(const float* __restrict__ W, _Float16* __restrict__ dst,
                       int Ksrc, int kt_base, int KT_total) {
  size_t tid = (size_t)blockIdx.x * 256 + threadIdx.x;   // N*Ksrc threads
  int j    = (int)(tid & 15);
  int lane = (int)((tid >> 4) & 31);
  size_t tl = tid >> 9;
  int KTsrc = Ksrc >> 5;
  int ktl = (int)(tl % KTsrc);
  int nt  = (int)(tl / KTsrc);
  int k = ktl * 32 + (lane >> 4) * 8 + (j & 7) + (j >> 3) * 16;
  int n = nt * 16 + (lane & 15);
  dst[(((size_t)nt * KT_total + kt_base + ktl) * 32 + lane) * 16 + j] =
      (_Float16)W[(size_t)n * Ksrc + k];
}

// ---------------------------------------------------------------------------
// generic WMMA GEMM: C(MxN) = A(f16 row-major) * Bpacked + bias
// one 16x16 output tile per wave, 8 waves per block
// ---------------------------------------------------------------------------
template <bool F16OUT>
__global__ void k_gemm(const _Float16* __restrict__ A, int lda,
                       const _Float16* __restrict__ Bp, int KT,
                       const float* __restrict__ bias,
                       void* __restrict__ Cout, int ldc, int ntiles_n) {
  int wave = threadIdx.x >> 5;
  int lane = threadIdx.x & 31;
  int tile = blockIdx.x * 8 + wave;
  int mt = tile / ntiles_n;
  int nt = tile - mt * ntiles_n;
  int m  = mt * 16 + (lane & 15);
  int kh = (lane >> 4) * 8;
  const _Float16* arow  = A + (size_t)m * lda + kh;
  const _Float16* bbase = Bp + (size_t)nt * KT * 512 + lane * 16;
  v8f acc = {};
  for (int kt = 0; kt < KT; ++kt) {
    v16h a = load_a_frag(arow, kt * 32);
    v16h b = *(const v16h*)(bbase + (size_t)kt * 512);
    acc = __builtin_amdgcn_wmma_f32_16x16x32_f16(false, a, false, b, (short)0, acc,
                                                 false, false);
  }
  int row = mt * 16 + (lane >> 4) * 8;
  int col = nt * 16 + (lane & 15);
  float bv = bias ? bias[col] : 0.0f;
#pragma unroll
  for (int r = 0; r < 8; ++r) {
    float vv = acc[r] + bv;
    if (F16OUT) ((_Float16*)Cout)[(size_t)(row + r) * ldc + col] = (_Float16)vv;
    else        ((float*)Cout)[(size_t)(row + r) * ldc + col] = vv;
  }
}

// ---------------------------------------------------------------------------
// projection GEMM: enc(4096x512) = [hf|hb](4096x1024) @ proj_w.T + proj_b (f16 out)
// A is selected per-kt from hf_seq (kt<16) or hb_seq (kt>=16)
// ---------------------------------------------------------------------------
__global__ void k_proj(const _Float16* __restrict__ hf, const _Float16* __restrict__ hb,
                       const _Float16* __restrict__ Bp, const float* __restrict__ bias,
                       _Float16* __restrict__ Cout) {
  const int KT = 32, ntiles_n = 32, ldc = Hh;
  int wave = threadIdx.x >> 5;
  int lane = threadIdx.x & 31;
  int tile = blockIdx.x * 8 + wave;
  int mt = tile / ntiles_n;
  int nt = tile - mt * ntiles_n;
  int m  = mt * 16 + (lane & 15);
  int kh = (lane >> 4) * 8;
  const _Float16* afrow = hf + (size_t)m * Hh + kh;
  const _Float16* abrow = hb + (size_t)m * Hh + kh;
  const _Float16* bbase = Bp + (size_t)nt * KT * 512 + lane * 16;
  v8f acc = {};
  for (int kt = 0; kt < KT; ++kt) {
    const _Float16* ar = (kt < 16) ? afrow : abrow;
    int koff = ((kt < 16) ? kt : (kt - 16)) * 32;
    v16h a = load_a_frag(ar, koff);
    v16h b = *(const v16h*)(bbase + (size_t)kt * 512);
    acc = __builtin_amdgcn_wmma_f32_16x16x32_f16(false, a, false, b, (short)0, acc,
                                                 false, false);
  }
  int row = mt * 16 + (lane >> 4) * 8;
  int col = nt * 16 + (lane & 15);
  float bv = bias[col];
#pragma unroll
  for (int r = 0; r < 8; ++r)
    Cout[(size_t)(row + r) * ldc + col] = (_Float16)(acc[r] + bv);
}

// ---------------------------------------------------------------------------
// init: c_f = c_b = c_dec = 0 ; Adec0 = [dec_start | 0] ; Adec1 = 0
// ---------------------------------------------------------------------------
__global__ void k_init(const float* __restrict__ dec_start, _Float16* __restrict__ A0,
                       _Float16* __restrict__ A1, float* __restrict__ c_dec,
                       float* __restrict__ c_f, float* __restrict__ c_b) {
  int tid = blockIdx.x * 256 + threadIdx.x;   // B*H
  int b = tid >> 9, k = tid & (Hh - 1);
  _Float16 z = (_Float16)0.0f;
  A0[(size_t)b * 1024 + k]        = (_Float16)dec_start[k];
  A0[(size_t)b * 1024 + Hh + k]   = z;
  A1[(size_t)b * 1024 + k]        = z;
  A1[(size_t)b * 1024 + Hh + k]   = z;
  c_dec[tid] = 0.0f;
  c_f[tid]   = 0.0f;
  c_b[tid]   = 0.0f;
}

// ---------------------------------------------------------------------------
// fused encoder LSTM step (both directions). 64 blocks:
//   blk<32  : forward,  column group j0 = blk*16,       t = s,     hprev row t-1
//   blk>=32 : backward, column group j0 = (blk-32)*16,  t = 127-s, hprev row t+1
// 8 waves: wave = gate*2 + mt ; gates GEMM (K=512) then gate nonlinearity.
// X already contains enc_in @ Wih.T + b.
// ---------------------------------------------------------------------------
__global__ void k_enc_step(const _Float16* __restrict__ Xf, const _Float16* __restrict__ Xb,
                           const _Float16* __restrict__ WhhF, const _Float16* __restrict__ WhhB,
                           float* __restrict__ c_f, float* __restrict__ c_b,
                           _Float16* __restrict__ hf_seq, _Float16* __restrict__ hb_seq,
                           int s) {
  __shared__ float lds[8 * 256];
  int blk = blockIdx.x;
  bool bwd = blk >= 32;
  int j0 = (bwd ? blk - 32 : blk) * 16;
  int t  = bwd ? (Ee - 1 - s) : s;
  int tprev = bwd ? (t + 1) : (t - 1);
  const _Float16* X   = bwd ? Xb : Xf;
  const _Float16* Wp  = bwd ? WhhB : WhhF;
  float* c            = bwd ? c_b : c_f;
  _Float16* hseq      = bwd ? hb_seq : hf_seq;

  int wave = threadIdx.x >> 5;
  int lane = threadIdx.x & 31;
  int mt = wave & 1;
  int gate = wave >> 1;
  int nt = gate * 32 + (j0 >> 4);
  v8f acc = {};
  if (s > 0) {
    int m  = mt * 16 + (lane & 15);
    int kh = (lane >> 4) * 8;
    const _Float16* arow  = hseq + ((size_t)(m * Ee + tprev)) * Hh + kh;
    const _Float16* bbase = Wp + (size_t)nt * 16 * 512 + lane * 16;
    for (int kt = 0; kt < 16; ++kt) {
      v16h a = load_a_frag(arow, kt * 32);
      v16h b = *(const v16h*)(bbase + (size_t)kt * 512);
      acc = __builtin_amdgcn_wmma_f32_16x16x32_f16(false, a, false, b, (short)0, acc,
                                                   false, false);
    }
  }
  int rbase = (lane >> 4) * 8;
  int colL  = lane & 15;
#pragma unroll
  for (int r = 0; r < 8; ++r) lds[wave * 256 + (rbase + r) * 16 + colL] = acc[r];
  __syncthreads();

  for (int q = threadIdx.x; q < 512; q += 256) {
    int brow = q >> 4;            // batch row 0..31
    int col  = q & 15;
    int mt2 = brow >> 4, r2 = brow & 15;
    int n = j0 + col;
    size_t xrow = (size_t)(brow * Ee + t) * H4;
    float gi = lds[(0 * 2 + mt2) * 256 + r2 * 16 + col] + (float)X[xrow + 0 * Hh + n];
    float gf = lds[(1 * 2 + mt2) * 256 + r2 * 16 + col] + (float)X[xrow + 1 * Hh + n];
    float gc = lds[(2 * 2 + mt2) * 256 + r2 * 16 + col] + (float)X[xrow + 2 * Hh + n];
    float go = lds[(3 * 2 + mt2) * 256 + r2 * 16 + col] + (float)X[xrow + 3 * Hh + n];
    float cn = sigm(gf) * c[brow * Hh + n] + sigm(gi) * tanhf(gc);
    float hn = sigm(go) * tanhf(cn);
    c[brow * Hh + n] = cn;
    hseq[((size_t)(brow * Ee + t)) * Hh + n] = (_Float16)hn;
  }
}

// ---------------------------------------------------------------------------
// decoder LSTMCell gates: g = Adec_r(32x1024) @ [dWih|dWhh].T + db -> h,c update
// writes hx (f16) into h-half of Adec_w. 32 blocks (16 columns each), 8 waves.
// ---------------------------------------------------------------------------
__global__ void k_dec_gates(const _Float16* __restrict__ Adec_r,
                            const _Float16* __restrict__ dWp, const float* __restrict__ db,
                            float* __restrict__ c_dec, _Float16* __restrict__ Adec_w) {
  __shared__ float lds[8 * 256];
  int j0 = blockIdx.x * 16;
  int wave = threadIdx.x >> 5;
  int lane = threadIdx.x & 31;
  int mt = wave & 1;
  int gate = wave >> 1;
  int nt = gate * 32 + (j0 >> 4);
  int m  = mt * 16 + (lane & 15);
  int kh = (lane >> 4) * 8;
  const _Float16* arow  = Adec_r + (size_t)m * 1024 + kh;
  const _Float16* bbase = dWp + (size_t)nt * 32 * 512 + lane * 16;
  v8f acc = {};
  for (int kt = 0; kt < 32; ++kt) {
    v16h a = load_a_frag(arow, kt * 32);
    v16h b = *(const v16h*)(bbase + (size_t)kt * 512);
    acc = __builtin_amdgcn_wmma_f32_16x16x32_f16(false, a, false, b, (short)0, acc,
                                                 false, false);
  }
  int rbase = (lane >> 4) * 8;
  int colL  = lane & 15;
#pragma unroll
  for (int r = 0; r < 8; ++r) lds[wave * 256 + (rbase + r) * 16 + colL] = acc[r];
  __syncthreads();

  for (int q = threadIdx.x; q < 512; q += 256) {
    int brow = q >> 4;
    int col  = q & 15;
    int mt2 = brow >> 4, r2 = brow & 15;
    int n = j0 + col;
    float gi = lds[(0 * 2 + mt2) * 256 + r2 * 16 + col] + db[0 * Hh + n];
    float gf = lds[(1 * 2 + mt2) * 256 + r2 * 16 + col] + db[1 * Hh + n];
    float gc = lds[(2 * 2 + mt2) * 256 + r2 * 16 + col] + db[2 * Hh + n];
    float go = lds[(3 * 2 + mt2) * 256 + r2 * 16 + col] + db[3 * Hh + n];
    float cn = sigm(gf) * c_dec[brow * Hh + n] + sigm(gi) * tanhf(gc);
    float hn = sigm(go) * tanhf(cn);
    c_dec[brow * Hh + n] = cn;
    Adec_w[(size_t)brow * 1024 + Hh + n] = (_Float16)hn;
  }
}

// ---------------------------------------------------------------------------
// attention + log_softmax + teacher-forced gather. 32 blocks (one per batch).
// u[e] = sum_k v[k]*tanh(enc_key[b,e,k]+dec_key[b,k]); scores=[u,0]; log_softmax
// then Adec_w x-half <- enc row target_seq[b,s] (or 0 if idx==E)
// ---------------------------------------------------------------------------
__global__ void k_attn(const float* __restrict__ enc_key, const float* __restrict__ dec_key,
                       const float* __restrict__ v, const _Float16* __restrict__ enc,
                       const unsigned char* __restrict__ mask, const int* __restrict__ tseq,
                       float* __restrict__ out, _Float16* __restrict__ Adec_w, int s) {
  __shared__ float s_dk[Hh];
  __shared__ float s_v[Hh];
  __shared__ float s_u[Ee + 1];
  __shared__ float s_red[2];
  int b = blockIdx.x;
  int tid = threadIdx.x;
  for (int k = tid; k < Hh; k += 256) {
    s_dk[k] = dec_key[b * Hh + k];
    s_v[k]  = v[k];
  }
  __syncthreads();
  int wave = tid >> 5, lane = tid & 31;
  for (int e = wave; e < Ee; e += 8) {
    const float* ek = enc_key + ((size_t)(b * Ee + e)) * Hh;
    float p = 0.0f;
    for (int k = lane; k < Hh; k += 32) p += s_v[k] * tanhf(ek[k] + s_dk[k]);
#pragma unroll
    for (int off = 16; off; off >>= 1) p += __shfl_xor(p, off, 32);
    if (lane == 0) s_u[e] = mask[b * Ee + e] ? p : -INFINITY;
  }
  if (tid == 0) s_u[Ee] = 0.0f;
  __syncthreads();
  if (tid == 0) {
    float m = -INFINITY;
    for (int j = 0; j <= Ee; ++j) m = fmaxf(m, s_u[j]);
    float sum = 0.0f;
    for (int j = 0; j <= Ee; ++j) sum += __expf(s_u[j] - m);
    s_red[0] = m;
    s_red[1] = __logf(sum);
  }
  __syncthreads();
  float m = s_red[0], l = s_red[1];
  for (int j = tid; j <= Ee; j += 256)
    out[((size_t)b * Tt + s) * (Ee + 1) + j] = s_u[j] - m - l;

  int idx = tseq[b * Tt + s];
  int ec = idx < 0 ? 0 : (idx > Ee - 1 ? Ee - 1 : idx);
  const _Float16* src = enc + ((size_t)(b * Ee + ec)) * Hh;
  _Float16 z = (_Float16)0.0f;
  for (int k = tid; k < Hh; k += 256)
    Adec_w[(size_t)b * 1024 + k] = (idx < Ee) ? src[k] : z;
}

// ---------------------------------------------------------------------------
// host launcher
// ---------------------------------------------------------------------------
extern "C" void kernel_launch(void* const* d_in, const int* in_sizes, int n_in,
                              void* d_out, int out_size, void* d_ws, size_t ws_size,
                              hipStream_t stream) {
  const float* endpoints = (const float*)d_in[0];
  const float* L         = (const float*)d_in[1];
  const float* ep_w      = (const float*)d_in[2];
  const float* ep_b      = (const float*)d_in[3];
  const float* Lw        = (const float*)d_in[4];
  const float* Lb        = (const float*)d_in[5];
  const float* Wih_f     = (const float*)d_in[6];
  const float* Whh_f     = (const float*)d_in[7];
  const float* b_f       = (const float*)d_in[8];
  const float* Wih_b     = (const float*)d_in[9];
  const float* Whh_b     = (const float*)d_in[10];
  const float* b_b       = (const float*)d_in[11];
  const float* proj_w    = (const float*)d_in[12];
  const float* proj_b    = (const float*)d_in[13];
  const float* dWih      = (const float*)d_in[14];
  const float* dWhh      = (const float*)d_in[15];
  const float* db        = (const float*)d_in[16];
  const float* dec_start = (const float*)d_in[17];
  const float* W_enc     = (const float*)d_in[18];
  const float* W_dec     = (const float*)d_in[19];
  const float* v         = (const float*)d_in[20];
  const unsigned char* mask = (const unsigned char*)d_in[21];
  const int* tseq        = (const int*)d_in[22];
  float* out             = (float*)d_out;

  // workspace carving (all sizes 256B-aligned already)
  char* base = (char*)d_ws;
  size_t o = 0;
  auto alloc = [&](size_t bytes) { void* p = base + o; o += (bytes + 255) & ~(size_t)255; return p; };
  _Float16* enc_in = (_Float16*)alloc((size_t)Bb * Ee * Hh * 2);
  _Float16* Xf     = (_Float16*)alloc((size_t)Bb * Ee * H4 * 2);
  _Float16* Xb     = (_Float16*)alloc((size_t)Bb * Ee * H4 * 2);
  _Float16* WihFp  = (_Float16*)alloc((size_t)H4 * Hh * 2);
  _Float16* WhhFp  = (_Float16*)alloc((size_t)H4 * Hh * 2);
  _Float16* WihBp  = (_Float16*)alloc((size_t)H4 * Hh * 2);
  _Float16* WhhBp  = (_Float16*)alloc((size_t)H4 * Hh * 2);
  _Float16* dWp    = (_Float16*)alloc((size_t)H4 * 1024 * 2);
  _Float16* projp  = (_Float16*)alloc((size_t)Hh * 1024 * 2);
  _Float16* Wencp  = (_Float16*)alloc((size_t)Hh * Hh * 2);
  _Float16* Wdecp  = (_Float16*)alloc((size_t)Hh * Hh * 2);
  _Float16* hf_seq = (_Float16*)alloc((size_t)Bb * Ee * Hh * 2);
  _Float16* hb_seq = (_Float16*)alloc((size_t)Bb * Ee * Hh * 2);
  _Float16* enc16  = (_Float16*)alloc((size_t)Bb * Ee * Hh * 2);
  float*    enckey = (float*)alloc((size_t)Bb * Ee * Hh * 4);
  _Float16* Adec0  = (_Float16*)alloc((size_t)Bb * 1024 * 2);
  _Float16* Adec1  = (_Float16*)alloc((size_t)Bb * 1024 * 2);
  float*    c_f    = (float*)alloc((size_t)Bb * Hh * 4);
  float*    c_b    = (float*)alloc((size_t)Bb * Hh * 4);
  float*    c_dec  = (float*)alloc((size_t)Bb * Hh * 4);
  float*    deckey = (float*)alloc((size_t)Bb * Hh * 4);
  (void)ws_size; (void)n_in; (void)in_sizes; (void)out_size;

  // 1) embedding
  k_embed<<<(Bb * Ee * Hh) / 256, 256, 0, stream>>>(endpoints, L, ep_w, ep_b, Lw, Lb, enc_in);

  // 2) pack all weights into WMMA B-fragment layout
  int g2048x512 = (H4 * Hh) / 256;
  int g512x1024 = (Hh * 1024) / 256;
  int g512x512  = (Hh * Hh) / 256;
  k_pack<<<g2048x512, 256, 0, stream>>>(Wih_f, WihFp, Hh, 0, 16);
  k_pack<<<g2048x512, 256, 0, stream>>>(Whh_f, WhhFp, Hh, 0, 16);
  k_pack<<<g2048x512, 256, 0, stream>>>(Wih_b, WihBp, Hh, 0, 16);
  k_pack<<<g2048x512, 256, 0, stream>>>(Whh_b, WhhBp, Hh, 0, 16);
  k_pack<<<g2048x512, 256, 0, stream>>>(dWih, dWp, Hh, 0, 32);
  k_pack<<<g2048x512, 256, 0, stream>>>(dWhh, dWp, Hh, 16, 32);
  k_pack<<<g512x1024, 256, 0, stream>>>(proj_w, projp, 1024, 0, 32);
  k_pack<<<g512x512, 256, 0, stream>>>(W_enc, Wencp, Hh, 0, 16);
  k_pack<<<g512x512, 256, 0, stream>>>(W_dec, Wdecp, Hh, 0, 16);

  // 3) state init
  k_init<<<(Bb * Hh) / 256, 256, 0, stream>>>(dec_start, Adec0, Adec1, c_dec, c_f, c_b);

  // 4) X = enc_in @ Wih.T + b  (both directions): 256x128 tiles -> 4096 blocks
  k_gemm<true><<<4096, 256, 0, stream>>>(enc_in, Hh, WihFp, 16, b_f, Xf, H4, 128);
  k_gemm<true><<<4096, 256, 0, stream>>>(enc_in, Hh, WihBp, 16, b_b, Xb, H4, 128);

  // 5) encoder recurrence: 128 sequential fused steps (fwd+bwd)
  for (int s = 0; s < Ee; ++s)
    k_enc_step<<<64, 256, 0, stream>>>(Xf, Xb, WhhFp, WhhBp, c_f, c_b, hf_seq, hb_seq, s);

  // 6) projection (f16 enc) and enc_key (f32)
  k_proj<<<1024, 256, 0, stream>>>(hf_seq, hb_seq, projp, proj_b, enc16);
  k_gemm<false><<<1024, 256, 0, stream>>>(enc16, Hh, Wencp, 16, nullptr, enckey, Hh, 32);

  // 7) decoder: 128 sequential steps, double-buffered [x|h] A-matrix
  for (int s = 0; s < Tt; ++s) {
    _Float16* Ar = (s & 1) ? Adec1 : Adec0;
    _Float16* Aw = (s & 1) ? Adec0 : Adec1;
    k_dec_gates<<<32, 256, 0, stream>>>(Ar, dWp, db, c_dec, Aw);
    k_gemm<false><<<8, 256, 0, stream>>>(Aw + Hh, 1024, Wdecp, 16, nullptr, deckey, Hh, 32);
    k_attn<<<32, 256, 0, stream>>>(enckey, deckey, v, enc16, mask, tseq, out, Aw, s);
  }
}
